// Wrap_Model_1_46712064311762
// MI455X (gfx1250) — matrix operand order repaired and checked by reading, not compiled, not run on hardware
//
#include <hip/hip_runtime.h>
#include <stdint.h>

// ---------------------------------------------------------------------------
// k-NN (k=2) squared-L2, gfx1250 wave32.
//   d2[m,n] = ||q||^2 + ||t||^2 - 2 q.t
// cross via bf16 hi/lo split: 3x v_wmma_f32_16x16x32_bf16 per 32-K step.
// Each block stages a 256-row train slab into LDS (converted ONCE), then
// sweeps all 128 query tiles.  ||q||^2 is constant per query, so it is
// dropped from the hot loop and added in phase 2 (doesn't affect top-2).
// ---------------------------------------------------------------------------

typedef __attribute__((ext_vector_type(16))) __bf16 v16bf;
typedef __attribute__((ext_vector_type(8)))  float  v8f;

#define DDIM       128
#define NTRAIN     200000
#define MTEST      2048
#define NROWS_BLK  256                      // train rows staged per block
#define NBLK       782                      // ceil(200000 / 256)
#define NQT        (MTEST / 16)             // 128 query tiles
#define LDS_STRIDE 136                      // u16 per row (128 + 8 pad) -> conflict-free

union Frag16 {
    v16bf          v;
    unsigned short h[16];
    unsigned int   w[8];
};

__device__ __forceinline__ unsigned short bf16_rne(float x) {
    unsigned u = __float_as_uint(x);
    u += 0x7FFFu + ((u >> 16) & 1u);
    return (unsigned short)(u >> 16);
}
__device__ __forceinline__ float bf16_to_f(unsigned short h) {
    return __uint_as_float(((unsigned)h) << 16);
}
__device__ __forceinline__ void split2(float x, unsigned short& h, unsigned short& l) {
    unsigned short hh = bf16_rne(x);
    h = hh;
    l = bf16_rne(x - bf16_to_f(hh));
}
// merge two sorted pairs (a1<=a2, o1<=o2) -> two smallest of the four.
// second = med3(a1, o1, min(a2,o2))  [valid since min(a2,o2) >= min(a1,o1)]
__device__ __forceinline__ void merge_top2(float& a1, float& a2, float o1, float o2) {
    float n2 = __builtin_amdgcn_fmed3f(a1, o1, fminf(a2, o2));
    float n1 = fminf(a1, o1);
    a1 = n1; a2 = n2;
}

// ---------------------------------------------------------------------------
// Pre-pass: convert queries to hi/lo bf16 (row-major, [128 hi][128 lo] per
// row = 512B) and compute ||q||^2.  One wave per row.
// ---------------------------------------------------------------------------
__global__ __launch_bounds__(256)
void convert_queries(const float* __restrict__ test,
                     unsigned short* __restrict__ qconv,
                     float* __restrict__ qsq) {
    const int wave = threadIdx.x >> 5;
    const int lane = threadIdx.x & 31;
    const int row  = blockIdx.x * 8 + wave;           // 8 blocks (MTEST/8/32) * 8 = 64 rows

    const float4 v = ((const float4*)(test + (size_t)row * DDIM))[lane];
    unsigned short h0, l0, h1, l1, h2, l2, h3, l3;
    split2(v.x, h0, l0); split2(v.y, h1, l1);
    split2(v.z, h2, l2); split2(v.w, h3, l3);

    uint2 H, L;
    H.x = (unsigned)h0 | ((unsigned)h1 << 16);
    H.y = (unsigned)h2 | ((unsigned)h3 << 16);
    L.x = (unsigned)l0 | ((unsigned)l1 << 16);
    L.y = (unsigned)l2 | ((unsigned)l3 << 16);

    unsigned short* rowp = qconv + (size_t)row * 256;
    ((uint2*)rowp)[lane]         = H;                 // hi half  [0..127]
    ((uint2*)(rowp + 128))[lane] = L;                 // lo half  [128..255]

    float sq = v.x * v.x + v.y * v.y + v.z * v.z + v.w * v.w;
    #pragma unroll
    for (int off = 16; off >= 1; off >>= 1) sq += __shfl_xor(sq, off);
    if (lane == 0) qsq[row] = sq;
}

// ---------------------------------------------------------------------------
// Main: block = 256-row train slab; converts slab to LDS once, sweeps all
// 128 query tiles with WMMA, writes per-(qtile, slab) top-2 partials of
// d' = ||t||^2 - 2 q.t   (d2 minus the per-query constant ||q||^2).
// ---------------------------------------------------------------------------
__global__ __launch_bounds__(256)
void knn_main(const float* __restrict__ train,
              const unsigned short* __restrict__ qconv,
              float* __restrict__ ws_top2) {
    __shared__ unsigned short lds_hi[NROWS_BLK * LDS_STRIDE];  // 69,632 B
    __shared__ unsigned short lds_lo[NROWS_BLK * LDS_STRIDE];  // 69,632 B
    __shared__ float          lds_tsq[NROWS_BLK];
    __shared__ float          red[2][8][16][2];                // double-buffered by qt parity

    const int blk  = blockIdx.x;
    const int tid  = threadIdx.x;
    const int wave = tid >> 5;
    const int lane = tid & 31;
    const int m16  = lane & 15;           // A row / B column / C column
    const int g    = lane >> 4;           // K-half select
    const float INF = __uint_as_float(0x7F800000u);

    // ---- stage + convert the slab (ONCE per block): thread t <-> row t ----
    {
        const long grow  = (long)blk * NROWS_BLK + tid;
        const bool vr    = grow < NTRAIN;
        const float4* src = (const float4*)(train + (size_t)(vr ? grow : 0) * DDIM);
        unsigned short* dh = lds_hi + tid * LDS_STRIDE;
        unsigned short* dl = lds_lo + tid * LDS_STRIDE;
        float sq = 0.f;
        #pragma unroll 1
        for (int i = 0; i < 32; ++i) {
            const float4 v = src[i];
            sq += v.x * v.x + v.y * v.y + v.z * v.z + v.w * v.w;
            unsigned short h0, l0, h1, l1, h2, l2, h3, l3;
            split2(v.x, h0, l0); split2(v.y, h1, l1);
            split2(v.z, h2, l2); split2(v.w, h3, l3);
            uint2 H, L;
            H.x = (unsigned)h0 | ((unsigned)h1 << 16);
            H.y = (unsigned)h2 | ((unsigned)h3 << 16);
            L.x = (unsigned)l0 | ((unsigned)l1 << 16);
            L.y = (unsigned)l2 | ((unsigned)l3 << 16);
            ((uint2*)dh)[i] = H;
            ((uint2*)dl)[i] = L;
        }
        lds_tsq[tid] = vr ? sq : INF;     // INF poisons tail-row distances
    }
    __syncthreads();

    // ---- sweep all query tiles ----
    #pragma unroll 1
    for (int qt = 0; qt < NQT; ++qt) {
        // A fragments straight from pre-converted queries (pure b128 loads).
        // Layout (A, contiguous k): VGPR n<8: K = 32s+16g+2n,+1 (B loaded the same way)
        Frag16 a_hi[4], a_lo[4];
        const unsigned short* qrow = qconv + (size_t)(qt * 16 + m16) * 256;
        #pragma unroll
        for (int s = 0; s < 4; ++s) {
            const int k0 = 32 * s + 16 * g;
            const int k1 = 32 * s + 16 * g + 8;
            *(uint4*)&a_hi[s].w[0] = *(const uint4*)(qrow + k0);
            *(uint4*)&a_hi[s].w[4] = *(const uint4*)(qrow + k1);
            *(uint4*)&a_lo[s].w[0] = *(const uint4*)(qrow + 128 + k0);
            *(uint4*)&a_lo[s].w[4] = *(const uint4*)(qrow + 128 + k1);
        }

        float b1[8], b2[8];
        #pragma unroll
        for (int r = 0; r < 8; ++r) { b1[r] = INF; b2[r] = INF; }

        #pragma unroll
        for (int it = 0; it < 2; ++it) {
            const int jl = it * 128 + wave * 16 + m16;     // LDS row (= B column)
            const unsigned short* bh = lds_hi + jl * LDS_STRIDE;
            const unsigned short* bl = lds_lo + jl * LDS_STRIDE;
            v8f acc = {0.f, 0.f, 0.f, 0.f, 0.f, 0.f, 0.f, 0.f};
            #pragma unroll
            for (int s = 0; s < 4; ++s) {
                const int kb = 32 * s + 16 * g;            // B: lane half -> K half
                Frag16 fh, fl;
                *(uint4*)&fh.w[0] = *(const uint4*)(bh + kb);
                *(uint4*)&fh.w[4] = *(const uint4*)(bh + kb + 8);
                *(uint4*)&fl.w[0] = *(const uint4*)(bl + kb);
                *(uint4*)&fl.w[4] = *(const uint4*)(bl + kb + 8);
                acc = __builtin_amdgcn_wmma_f32_16x16x32_bf16(
                          false, a_hi[s].v, false, fh.v, (short)0, acc, false, false);
                acc = __builtin_amdgcn_wmma_f32_16x16x32_bf16(
                          false, a_lo[s].v, false, fh.v, (short)0, acc, false, false);
                acc = __builtin_amdgcn_wmma_f32_16x16x32_bf16(
                          false, a_hi[s].v, false, fl.v, (short)0, acc, false, false);
            }
            const float tsq = lds_tsq[jl];                 // +INF on tail rows
            #pragma unroll
            for (int r = 0; r < 8; ++r) {
                const float d = fmaf(-2.0f, acc[r], tsq);  // d2 - ||q||^2
                // sorted insert into (b1,b2): second-smallest = med3(b1,b2,d)
                const float n2 = __builtin_amdgcn_fmed3f(b1[r], b2[r], d);
                b1[r] = fminf(b1[r], d);
                b2[r] = n2;
            }
        }

        // reduce top-2 across the 16 N-lanes of each half-wave
        #pragma unroll
        for (int r = 0; r < 8; ++r) {
            #pragma unroll
            for (int off = 8; off >= 1; off >>= 1) {
                const float o1 = __shfl_xor(b1[r], off);
                const float o2 = __shfl_xor(b2[r], off);
                merge_top2(b1[r], b2[r], o1, o2);
            }
        }
        const int pp = qt & 1;
        if (m16 == 0) {
            #pragma unroll
            for (int r = 0; r < 8; ++r) {
                red[pp][wave][8 * g + r][0] = b1[r];
                red[pp][wave][8 * g + r][1] = b2[r];
            }
        }
        __syncthreads();   // single barrier/qt; parity buffer protects reads
        if (tid < 16) {
            float f1 = red[pp][0][tid][0];
            float f2 = red[pp][0][tid][1];
            #pragma unroll
            for (int w = 1; w < 8; ++w)
                merge_top2(f1, f2, red[pp][w][tid][0], red[pp][w][tid][1]);
            const size_t idx = ((size_t)qt * NBLK + blk) * 32 + (size_t)tid * 2;
            ws_top2[idx]     = f1;
            ws_top2[idx + 1] = f2;
        }
    }
}

// ---------------------------------------------------------------------------
// Phase 2: merge the NBLK per-slab top-2 partials per query;
// out = f1 + f2 + 2*||q||^2   (re-adding the deferred per-query constant).
// ---------------------------------------------------------------------------
__global__ __launch_bounds__(256)
void knn_top2_final(const float* __restrict__ ws_top2,
                    const float* __restrict__ qsq,
                    float* __restrict__ out) {
    const int m = blockIdx.x * blockDim.x + threadIdx.x;
    if (m >= MTEST) return;
    const int qt = m >> 4;
    const int q  = m & 15;
    const float* base = ws_top2 + (size_t)qt * NBLK * 32 + (size_t)q * 2;
    float f1 = __uint_as_float(0x7F800000u), f2 = f1;
    #pragma unroll 1
    for (int c = 0; c < NBLK; ++c) {
        merge_top2(f1, f2, base[(size_t)c * 32], base[(size_t)c * 32 + 1]);
    }
    out[m] = f1 + f2 + 2.0f * qsq[m];
}

// ---------------------------------------------------------------------------
// Workspace layout (bytes):
//   [0)          qconv : 2048 * 256 u16      = 1,048,576
//   [1,048,576)  qsq   : 2048 f32            =     8,192
//   [1,056,768)  top2  : 128*782*16*2 f32    = 12,812,288
//   total ~13.9 MB
// ---------------------------------------------------------------------------
extern "C" void kernel_launch(void* const* d_in, const int* in_sizes, int n_in,
                              void* d_out, int out_size, void* d_ws, size_t ws_size,
                              hipStream_t stream) {
    const float* train = (const float*)d_in[0];   // [200000, 128] f32
    const float* test  = (const float*)d_in[1];   // [2048, 128]  f32
    float*       out   = (float*)d_out;           // [2048] f32

    unsigned short* qconv = (unsigned short*)d_ws;
    float* qsq  = (float*)((char*)d_ws + 1048576);
    float* top2 = (float*)((char*)d_ws + 1056768);

    convert_queries<<<dim3(MTEST / 8 / 32), dim3(256), 0, stream>>>(test, qconv, qsq);
    knn_main<<<dim3(NBLK), dim3(256), 0, stream>>>(train, qconv, top2);
    knn_top2_final<<<dim3((MTEST + 255) / 256), dim3(256), 0, stream>>>(top2, qsq, out);
}
